// MoIETransformerBlock_40870908788983
// MI455X (gfx1250) — matrix operand order repaired and checked by reading, not compiled, not run on hardware
//
#include <hip/hip_runtime.h>

// MoIE transformer block for MI455X (gfx1250, wave32, WMMA).
// B=4, S=2048, D=1024. All GEMMs via v_wmma_f32_16x16x32_bf16 (bf16 in, fp32 acc).
// Projection GEMMs: B operand staged to LDS with global_load_async_to_lds_b128,
// double-buffered on ASYNCcnt (64-wide K chunks, 36KB LDS). ~192MB workspace.

#define BB   4
#define SS   2048
#define DD   1024
#define ROWS (BB * SS)   // 8192 flattened rows
#define KC   64          // K-chunk staged per double-buffer stage
#define LP   72          // LDS row pitch in bf16 elems (144B: 16B-aligned, bank-spread)

typedef __bf16 bf16;
typedef __attribute__((ext_vector_type(8)))  bf16  v8bf;
typedef __attribute__((ext_vector_type(16))) bf16  v16bf;
typedef __attribute__((ext_vector_type(8)))  float v8f;

union BF16x16 { v16bf v; v8bf h[2]; };

__device__ __forceinline__ unsigned short f2bf(float f) {
    unsigned int u = __float_as_uint(f);
    u += 0x7FFFu + ((u >> 16) & 1u);          // round-to-nearest-even
    return (unsigned short)(u >> 16);
}
__device__ __forceinline__ float bf2f(unsigned int s) {
    return __uint_as_float(s << 16);
}
__device__ __forceinline__ v8f wmma_bf16(v16bf a, v16bf b, v8f c) {
    return __builtin_amdgcn_wmma_f32_16x16x32_bf16(false, a, false, b, (short)0, c, false, false);
}
// Async copy 16B global -> LDS (per lane), tracked by ASYNCcnt.
__device__ __forceinline__ void async_g2l_b128(unsigned lds_addr, const void* gptr) {
    asm volatile("global_load_async_to_lds_b128 %0, %1, off"
                 :: "v"(lds_addr), "v"((unsigned long long)(uintptr_t)gptr)
                 : "memory");
}
__device__ __forceinline__ void wait_asynccnt0() {
    asm volatile("s_wait_asynccnt 0x0" ::: "memory");
}

// ---------------------------------------------------------------- LayerNorm
__global__ void ln_kernel(const float* __restrict__ x, const float* __restrict__ g,
                          const float* __restrict__ b, float* __restrict__ ln32,
                          unsigned short* __restrict__ ln16) {
    size_t row = blockIdx.x;
    int t = threadIdx.x;
    const float* xr = x + row * (size_t)DD;
    int i = t * 4;
    float4 v = *reinterpret_cast<const float4*>(xr + i);
    float s  = v.x + v.y + v.z + v.w;
    float s2 = v.x*v.x + v.y*v.y + v.z*v.z + v.w*v.w;
    __shared__ float sm[256], sq[256];
    sm[t] = s; sq[t] = s2; __syncthreads();
    for (int o = 128; o > 0; o >>= 1) {
        if (t < o) { sm[t] += sm[t + o]; sq[t] += sq[t + o]; }
        __syncthreads();
    }
    float mean = sm[0] * (1.0f / DD);
    float var  = sq[0] * (1.0f / DD) - mean * mean;
    float rs   = rsqrtf(var + 1e-5f);
    float4 gv = *reinterpret_cast<const float4*>(g + i);
    float4 bv = *reinterpret_cast<const float4*>(b + i);
    float4 y;
    y.x = (v.x - mean) * rs * gv.x + bv.x;
    y.y = (v.y - mean) * rs * gv.y + bv.y;
    y.z = (v.z - mean) * rs * gv.z + bv.z;
    y.w = (v.w - mean) * rs * gv.w + bv.w;
    *reinterpret_cast<float4*>(ln32 + row * (size_t)DD + i) = y;
    uint2 p;
    p.x = (unsigned)f2bf(y.x) | ((unsigned)f2bf(y.y) << 16);
    p.y = (unsigned)f2bf(y.z) | ((unsigned)f2bf(y.w) << 16);
    *reinterpret_cast<uint2*>(ln16 + row * (size_t)DD + i) = p;
}

// ----------------------------------------------------- fp32 -> bf16 convert
__global__ void conv_bf16_kernel(const float* __restrict__ src,
                                 unsigned short* __restrict__ dst, int n) {
    int i = (blockIdx.x * blockDim.x + threadIdx.x) * 4;
    if (i >= n) return;
    float4 v = *reinterpret_cast<const float4*>(src + i);
    uint2 p;
    p.x = (unsigned)f2bf(v.x) | ((unsigned)f2bf(v.y) << 16);
    p.y = (unsigned)f2bf(v.z) | ((unsigned)f2bf(v.w) << 16);
    *reinterpret_cast<uint2*>(dst + i) = p;
}

// ------------------------------------------------ gate -> cost = g/(max|g|+eps)
__global__ void cost_kernel(const float* g0, const float* g1, const float* g2,
                            const float* g3, float* __restrict__ cost) {
    const float* gp = (blockIdx.x == 0) ? g0 : (blockIdx.x == 1) ? g1
                     : (blockIdx.x == 2) ? g2 : g3;
    int t = threadIdx.x;
    float m = 0.0f;
#pragma unroll
    for (int j = 0; j < 4; j++) m = fmaxf(m, fabsf(gp[t + 256 * j]));
    __shared__ float sm[256];
    sm[t] = m; __syncthreads();
    for (int o = 128; o > 0; o >>= 1) {
        if (t < o) sm[t] = fmaxf(sm[t], sm[t + o]);
        __syncthreads();
    }
    float inv = 1.0f / (sm[0] + 1e-9f);
#pragma unroll
    for (int j = 0; j < 4; j++)
        cost[(size_t)blockIdx.x * DD + t + 256 * j] = gp[t + 256 * j] * inv;
}

// -------------------------------------------- fused dual-GEMM SPL + branch
// Block = 8 waves sharing one 64-col N-tile; each wave owns 16 M rows
// (8 fp32 accumulators per wave -> no spills). Both weight matrices staged
// 64x64 bf16 chunks into LDS via async copies, double-buffered on ASYNCcnt.
// mode 0: bf16 row-major out (q/k); mode 1: bf16 transposed (v -> vT[b][d][s]);
// mode 2: fp32 out with residual add (o-branch -> d_out).
__global__ void __launch_bounds__(256)
spl_branch_kernel(const unsigned short* __restrict__ A16,
                  const unsigned short* __restrict__ Wmu,
                  const unsigned short* __restrict__ Wpr,
                  const float* __restrict__ mu_b,
                  const float* __restrict__ cost,
                  const float* __restrict__ pass32,
                  int mode,
                  unsigned short* __restrict__ out16,
                  float* __restrict__ out32,
                  const float* __restrict__ resid) {
    __shared__ unsigned short ldsB[2][2][64 * LP];   // [buf][mu/pr][row*LP+k]

    int lane = threadIdx.x & 31;
    int wid  = threadIdx.x >> 5;                     // 0..7
    int l = lane & 15, h = lane >> 4;
    int n0 = blockIdx.x * 64;
    int m0 = blockIdx.y * 128 + wid * 16;

    int sr   = threadIdx.x >> 2;                     // staging row 0..63
    int sseg = threadIdx.x & 3;                      // 16B segment 0..3

    // Per-thread staging pointers (advance by KC elems per chunk).
    const unsigned short* gmu = Wmu + (size_t)(n0 + sr) * DD + sseg * 8;
    const unsigned short* gpr = Wpr + (size_t)(n0 + sr) * DD + sseg * 8;
    const unsigned dm[2] = {
        (unsigned)(uintptr_t)&ldsB[0][0][sr * LP + sseg * 8],
        (unsigned)(uintptr_t)&ldsB[1][0][sr * LP + sseg * 8]};
    const unsigned dp[2] = {
        (unsigned)(uintptr_t)&ldsB[0][1][sr * LP + sseg * 8],
        (unsigned)(uintptr_t)&ldsB[1][1][sr * LP + sseg * 8]};

    v8f accM[4] = {};
    v8f accP[4] = {};
    const unsigned short* Arow = A16 + (size_t)(m0 + l) * DD + h * 16;

    // Each thread copies 2x16B per matrix per stage (cols sseg*8 and sseg*8+32).
    auto stage = [&](int buf) {
        async_g2l_b128(dm[buf],      gmu);
        async_g2l_b128(dm[buf] + 64, gmu + 32);
        async_g2l_b128(dp[buf],      gpr);
        async_g2l_b128(dp[buf] + 64, gpr + 32);
        gmu += KC; gpr += KC;
    };

    const int NK = DD / KC;                          // 16 chunks
    stage(0);
    for (int kc = 0; kc < NK; ++kc) {
        int buf = kc & 1;
        wait_asynccnt0();                            // this wave's stage done
        __syncthreads();                             // all waves' stage visible
        if (kc + 1 < NK) stage(buf ^ 1);

        const unsigned short* mbase = &ldsB[buf][0][0];
        const unsigned short* pbase = &ldsB[buf][1][0];
        int k0 = kc * KC;
#pragma unroll
        for (int ks = 0; ks < KC; ks += 32) {
            BF16x16 a;
            a.h[0] = *reinterpret_cast<const v8bf*>(Arow + k0 + ks);
            a.h[1] = *reinterpret_cast<const v8bf*>(Arow + k0 + ks + 8);
#pragma unroll
            for (int t = 0; t < 4; t++) {
                int ro = (t * 16 + l) * LP + ks + h * 16;
                BF16x16 bm, bp;
                bm.h[0] = *reinterpret_cast<const v8bf*>(mbase + ro);
                bm.h[1] = *reinterpret_cast<const v8bf*>(mbase + ro + 8);
                bp.h[0] = *reinterpret_cast<const v8bf*>(pbase + ro);
                bp.h[1] = *reinterpret_cast<const v8bf*>(pbase + ro + 8);
                accM[t] = wmma_bf16(a.v, bm.v, accM[t]);
                accP[t] = wmma_bf16(a.v, bp.v, accP[t]);
            }
        }
    }

#pragma unroll
    for (int t = 0; t < 4; t++) {
        int col = n0 + t * 16 + l;
        float cc = cost[col];
        float bi = mu_b[col];
#pragma unroll
        for (int r = 0; r < 8; r++) {
            int row = m0 + h * 8 + r;
            size_t idx = (size_t)row * DD + col;
            float zm   = accM[t][r] + bi;
            float comp = zm / (1.0f + __expf(-zm));      // silu
            float rl   = accP[t][r] * 0.03125f - cc;     // match/sqrt(D)-cost
            float res  = (rl > 0.0f) ? (comp * rl) : pass32[idx];
            if (mode == 0) {
                out16[idx] = f2bf(res);
            } else if (mode == 1) {
                int bidx = row >> 11;                    // row / S
                int s    = row & (SS - 1);
                out16[((size_t)bidx * DD + col) * SS + s] = f2bf(res);
            } else {
                out32[idx] = resid[idx] + res;
            }
        }
    }
}

// ------------------------------------------------- scores = q k^T / sqrt(D)
// Wave computes 32 q-rows x 64 k-cols (2 A frags share each B frag).
__global__ void __launch_bounds__(256)
scores_kernel(const unsigned short* __restrict__ q16,
              const unsigned short* __restrict__ k16,
              unsigned short* __restrict__ sc) {
    const int MT = SS / 32, NT = SS / 64;            // 64, 32
    int wave = (blockIdx.x * blockDim.x + threadIdx.x) >> 5;
    int lane = threadIdx.x & 31;
    int b   = wave / (MT * NT);
    int rem = wave % (MT * NT);
    int mt = rem / NT, nt = rem % NT;
    int m0 = mt * 32, n0 = nt * 64;
    int l = lane & 15, h = lane >> 4;
    unsigned short* out = sc + (size_t)b * SS * SS;

    if (n0 > m0 + 31) {                              // fully masked tile
        unsigned short neg = f2bf(-1e9f);
#pragma unroll
        for (int ms = 0; ms < 2; ms++)
#pragma unroll
            for (int t = 0; t < 4; t++) {
                int col = n0 + t * 16 + l;
#pragma unroll
                for (int r = 0; r < 8; r++)
                    out[(size_t)(m0 + ms * 16 + h * 8 + r) * SS + col] = neg;
            }
        return;
    }

    v8f acc[2][4] = {};
    const unsigned short* Arow0 = q16 + (size_t)(b * SS + m0 + l) * DD + h * 16;
    const unsigned short* Arow1 = q16 + (size_t)(b * SS + m0 + 16 + l) * DD + h * 16;
    const unsigned short* Brow  = k16 + (size_t)(b * SS + n0 + l) * DD + h * 16;
    for (int k0 = 0; k0 < DD; k0 += 32) {
        BF16x16 a0, a1;
        a0.h[0] = *reinterpret_cast<const v8bf*>(Arow0 + k0);
        a0.h[1] = *reinterpret_cast<const v8bf*>(Arow0 + k0 + 8);
        a1.h[0] = *reinterpret_cast<const v8bf*>(Arow1 + k0);
        a1.h[1] = *reinterpret_cast<const v8bf*>(Arow1 + k0 + 8);
#pragma unroll
        for (int t = 0; t < 4; t++) {
            v16bf bk = *reinterpret_cast<const v16bf*>(Brow + (size_t)t * 16 * DD + k0);
            acc[0][t] = wmma_bf16(a0.v, bk, acc[0][t]);
            acc[1][t] = wmma_bf16(a1.v, bk, acc[1][t]);
        }
    }
#pragma unroll
    for (int ms = 0; ms < 2; ms++)
#pragma unroll
        for (int t = 0; t < 4; t++) {
            int col = n0 + t * 16 + l;
#pragma unroll
            for (int r = 0; r < 8; r++) {
                int row = m0 + ms * 16 + h * 8 + r;
                float v = acc[ms][t][r] * 0.03125f;
                if (col > row) v = -1e9f;
                out[(size_t)row * SS + col] = f2bf(v);
            }
        }
}

// ---------------------------------------------- in-place row softmax (bf16)
__global__ void softmax_kernel(unsigned short* __restrict__ sc) {
    size_t row = blockIdx.x;
    int t = threadIdx.x;
    unsigned short* rp = sc + row * (size_t)SS + t * 8;
    uint4 raw = *reinterpret_cast<const uint4*>(rp);
    unsigned int w[4] = {raw.x, raw.y, raw.z, raw.w};
    float v[8];
#pragma unroll
    for (int j = 0; j < 4; j++) {
        v[2 * j]     = bf2f(w[j] & 0xFFFFu);
        v[2 * j + 1] = __uint_as_float(w[j] & 0xFFFF0000u);
    }
    float m = v[0];
#pragma unroll
    for (int j = 1; j < 8; j++) m = fmaxf(m, v[j]);
    __shared__ float sm[256];
    sm[t] = m; __syncthreads();
    for (int o = 128; o > 0; o >>= 1) {
        if (t < o) sm[t] = fmaxf(sm[t], sm[t + o]);
        __syncthreads();
    }
    m = sm[0]; __syncthreads();
    float e[8], s = 0.0f;
#pragma unroll
    for (int j = 0; j < 8; j++) { e[j] = __expf(v[j] - m); s += e[j]; }
    sm[t] = s; __syncthreads();
    for (int o = 128; o > 0; o >>= 1) {
        if (t < o) sm[t] += sm[t + o];
        __syncthreads();
    }
    float inv = 1.0f / sm[0];
    uint4 po;
    po.x = (unsigned)f2bf(e[0] * inv) | ((unsigned)f2bf(e[1] * inv) << 16);
    po.y = (unsigned)f2bf(e[2] * inv) | ((unsigned)f2bf(e[3] * inv) << 16);
    po.z = (unsigned)f2bf(e[4] * inv) | ((unsigned)f2bf(e[5] * inv) << 16);
    po.w = (unsigned)f2bf(e[6] * inv) | ((unsigned)f2bf(e[7] * inv) << 16);
    *reinterpret_cast<uint4*>(rp) = po;
}

// ------------------------------------- attn = p @ v (causally trimmed K loop)
__global__ void __launch_bounds__(256)
pv_kernel(const unsigned short* __restrict__ p16,
          const unsigned short* __restrict__ vT16,
          float* __restrict__ attn32,
          unsigned short* __restrict__ attn16) {
    const int MT = SS / 32, NT = DD / 64;            // 64, 16
    int wave = (blockIdx.x * blockDim.x + threadIdx.x) >> 5;
    int lane = threadIdx.x & 31;
    int b   = wave / (MT * NT);
    int rem = wave % (MT * NT);
    int mt = rem / NT, nt = rem % NT;
    int m0 = mt * 32, n0 = nt * 64;
    int l = lane & 15, h = lane >> 4;
    int kmax = m0 + 32;                              // past-diagonal p == 0

    v8f acc[2][4] = {};
    const unsigned short* Arow0 = p16  + ((size_t)b * SS + m0 + l) * SS + h * 16;
    const unsigned short* Arow1 = p16  + ((size_t)b * SS + m0 + 16 + l) * SS + h * 16;
    const unsigned short* Brow  = vT16 + ((size_t)b * DD + n0 + l) * SS + h * 16;
    for (int k0 = 0; k0 < kmax; k0 += 32) {
        BF16x16 a0, a1;
        a0.h[0] = *reinterpret_cast<const v8bf*>(Arow0 + k0);
        a0.h[1] = *reinterpret_cast<const v8bf*>(Arow0 + k0 + 8);
        a1.h[0] = *reinterpret_cast<const v8bf*>(Arow1 + k0);
        a1.h[1] = *reinterpret_cast<const v8bf*>(Arow1 + k0 + 8);
#pragma unroll
        for (int t = 0; t < 4; t++) {
            v16bf bv = *reinterpret_cast<const v16bf*>(Brow + (size_t)t * 16 * SS + k0);
            acc[0][t] = wmma_bf16(a0.v, bv, acc[0][t]);
            acc[1][t] = wmma_bf16(a1.v, bv, acc[1][t]);
        }
    }
#pragma unroll
    for (int ms = 0; ms < 2; ms++)
#pragma unroll
        for (int t = 0; t < 4; t++) {
            int col = n0 + t * 16 + l;
#pragma unroll
            for (int r = 0; r < 8; r++) {
                int row = m0 + ms * 16 + h * 8 + r;
                size_t idx = ((size_t)b * SS + row) * DD + col;
                float v = acc[ms][t][r];
                attn32[idx] = v;
                attn16[idx] = f2bf(v);
            }
        }
}

// -----------------------------------------------------------------------------
extern "C" void kernel_launch(void* const* d_in, const int* in_sizes, int n_in,
                              void* d_out, int out_size, void* d_ws, size_t ws_size,
                              hipStream_t stream) {
    const float* x    = (const float*)d_in[0];
    const float* ln_g = (const float*)d_in[1];
    const float* ln_b = (const float*)d_in[2];
    const float* mu_w[4]  = {(const float*)d_in[3],  (const float*)d_in[7],
                             (const float*)d_in[11], (const float*)d_in[15]};
    const float* mu_b[4]  = {(const float*)d_in[4],  (const float*)d_in[8],
                             (const float*)d_in[12], (const float*)d_in[16]};
    const float* proto[4] = {(const float*)d_in[5],  (const float*)d_in[9],
                             (const float*)d_in[13], (const float*)d_in[17]};
    const float* gate[4]  = {(const float*)d_in[6],  (const float*)d_in[10],
                             (const float*)d_in[14], (const float*)d_in[18]};

    char* ws = (char*)d_ws;
    size_t off = 0;
    auto take = [&](size_t bytes) -> char* {
        char* p = ws + off;
        off += (bytes + 255) & ~(size_t)255;
        return p;
    };

    float*          ln32   = (float*)take((size_t)ROWS * DD * 4);
    unsigned short* ln16   = (unsigned short*)take((size_t)ROWS * DD * 2);
    unsigned short* w16[8];
    for (int i = 0; i < 8; i++) w16[i] = (unsigned short*)take((size_t)DD * DD * 2);
    float*          cost   = (float*)take((size_t)4 * DD * 4);
    unsigned short* q16    = (unsigned short*)take((size_t)ROWS * DD * 2);
    unsigned short* k16    = (unsigned short*)take((size_t)ROWS * DD * 2);
    unsigned short* vT16   = (unsigned short*)take((size_t)ROWS * DD * 2);
    unsigned short* sc16   = (unsigned short*)take((size_t)BB * SS * SS * 2);
    float*          attn32 = (float*)take((size_t)ROWS * DD * 4);
    unsigned short* attn16 = (unsigned short*)take((size_t)ROWS * DD * 2);
    (void)ws_size; (void)in_sizes; (void)n_in; (void)out_size;

    ln_kernel<<<ROWS, 256, 0, stream>>>(x, ln_g, ln_b, ln32, ln16);

    const int convBlocks = (DD * DD / 4 + 255) / 256;        // 1024
    for (int p = 0; p < 4; p++) {
        conv_bf16_kernel<<<convBlocks, 256, 0, stream>>>(mu_w[p],  w16[2 * p],     DD * DD);
        conv_bf16_kernel<<<convBlocks, 256, 0, stream>>>(proto[p], w16[2 * p + 1], DD * DD);
    }
    cost_kernel<<<4, 256, 0, stream>>>(gate[0], gate[1], gate[2], gate[3], cost);

    dim3 splGrid(DD / 64, ROWS / 128);                       // (16, 64)
    spl_branch_kernel<<<splGrid, 256, 0, stream>>>(ln16, w16[0], w16[1], mu_b[0],
                                                   cost + 0 * DD, ln32, 0, q16, nullptr, nullptr);
    spl_branch_kernel<<<splGrid, 256, 0, stream>>>(ln16, w16[2], w16[3], mu_b[1],
                                                   cost + 1 * DD, ln32, 0, k16, nullptr, nullptr);
    spl_branch_kernel<<<splGrid, 256, 0, stream>>>(ln16, w16[4], w16[5], mu_b[2],
                                                   cost + 2 * DD, ln32, 1, vT16, nullptr, nullptr);

    const int scBlocks = BB * (SS / 32) * (SS / 64) / 8;     // 1024
    scores_kernel<<<scBlocks, 256, 0, stream>>>(q16, k16, sc16);
    softmax_kernel<<<ROWS, 256, 0, stream>>>(sc16);
    const int pvBlocks = BB * (SS / 32) * (DD / 64) / 8;     // 512
    pv_kernel<<<pvBlocks, 256, 0, stream>>>(sc16, vT16, attn32, attn16);

    spl_branch_kernel<<<splGrid, 256, 0, stream>>>(attn16, w16[6], w16[7], mu_b[3],
                                                   cost + 3 * DD, attn32, 2, nullptr,
                                                   (float*)d_out, x);
}
